// ClassGenMPS_61168924230445
// MI455X (gfx1250) — matrix operand order repaired and checked by reading, not compiled, not run on hardware
//
#include <hip/hip_runtime.h>
#include <hip/hip_bf16.h>
#include <stdint.h>

// MPS classifier chain for MI455X (gfx1250, wave32, WMMA bf16 16x16x32).
//
// Strategy:
//  - A[n,b,l,r] (256MB if materialized) is never formed. Each chain step is
//    recast as a GEMM: U[b,(l,k)] = Al[b,l]*x[i,b,k]  (B=256 x K=128 outer
//    product, built in VALU) times T[i][(l,k),r] (128x32), accumulated in f32
//    by v_wmma_f32_16x16x32_bf16.
//  - Batch rows are independent across the whole chain: each wave owns a
//    16-row x 32-col Al (or Ar) slice for all 127 steps, no cross-wave sync.
//    16 waves do the left chain, 16 the right chain, concurrently.
//  - A prep kernel pre-swizzles `tensor` (f32) into bf16 tables in the exact
//    WMMA B-operand register layout, so the chain's critical path loads each
//    B operand with contiguous 128-bit loads (L2-resident, 2MB/table).
//  - The chain loop is software-pipelined: site s+1's B operands are loaded
//    (ping-pong double buffer, 2-step unroll so the swap is register
//    renaming) while site s computes, and site s+2 is prefetched
//    (global_prefetch_b8) to pre-warm the WGP cache. This hides the full L2
//    round trip on the serialized critical path.
//
// Workspace layout (needs ~4.07 MB):
//   [0, 2MB)            TLs: left-chain B tables, bf16, WMMA-swizzled
//   [2MB, 4MB)          TRs: right-chain B tables
//   [4MB, 4MB+32KB)     AlBuf: final left environments  [B=256][D=32] f32
//   [4MB+32KB, +64KB)   ArBuf: final right environments [B=256][D=32] f32

typedef __attribute__((ext_vector_type(16))) __bf16 v16bf;
typedef __attribute__((ext_vector_type(8)))  float  v8f;

#define N_SITES 256
#define BATCH   256
#define DLOC    4
#define DB      32
#define NCLS    10

static __device__ __forceinline__ unsigned bf16bits(float f) {
    union { float f; unsigned u; } v; v.f = f;
    // round-to-nearest-even f32 -> bf16 (prep kernel only; off critical path)
    return (v.u + 0x7FFFu + ((v.u >> 16) & 1u)) >> 16;
}

// ---------------------------------------------------------------------------
// Prep: tensor[f32 N,D,D,d] -> two bf16 tables in WMMA B-operand layout.
// B 32x16 bf16 layout: lane = 16*half + n (col N = n), dword j of lane holds
// K = 32*c + half*16 + 2*j (lo) and K+1 (hi), for chunk c, col-tile t.
// Left  table: K encodes (l,k) = (K>>2, K&3), col = r.
// Right table: K encodes (r,k) = (K>>2, K&3), col = l.
// ---------------------------------------------------------------------------
__global__ __launch_bounds__(256) void mps_prep(const float* __restrict__ tensor,
                                                unsigned* __restrict__ TLs,
                                                unsigned* __restrict__ TRs) {
    unsigned tid = blockIdx.x * 256u + threadIdx.x;
    unsigned table = tid >> 19;           // 2 tables x 2^19 dwords
    unsigned rem   = tid & 0x7FFFFu;
    unsigned j    = rem & 7u;
    unsigned lane = (rem >> 3) & 31u;
    unsigned t    = (rem >> 8) & 1u;
    unsigned c    = (rem >> 9) & 3u;
    unsigned i    = rem >> 11;            // site
    unsigned half = lane >> 4, n = lane & 15u;
    unsigned K0 = 32u * c + half * 16u + 2u * j;
    unsigned K1 = K0 + 1u;
    float v0, v1;
    if (table == 0) {
        unsigned r = 16u * t + n;
        v0 = tensor[((i * DB + (K0 >> 2)) * DB + r) * DLOC + (K0 & 3u)];
        v1 = tensor[((i * DB + (K1 >> 2)) * DB + r) * DLOC + (K1 & 3u)];
    } else {
        unsigned lc = 16u * t + n;
        v0 = tensor[((i * DB + lc) * DB + (K0 >> 2)) * DLOC + (K0 & 3u)];
        v1 = tensor[((i * DB + lc) * DB + (K1 >> 2)) * DLOC + (K1 & 3u)];
    }
    unsigned dw = bf16bits(v0) | (bf16bits(v1) << 16);
    unsigned idx = (((i * 4u + c) * 2u + t) * 32u + lane) * 8u + j;
    (table == 0 ? TLs : TRs)[idx] = dw;
}

// ---------------------------------------------------------------------------
// Chain: blocks 0..15 run the left chain (batch tile = blockIdx), blocks
// 16..31 run the right chain. One wave (32 threads) per block; each wave owns
// a 16x32 environment slice in WMMA C/D layout: acc[t][v] = env[row = v+8*half]
// [col = 16*t + (lane&15)].
// ---------------------------------------------------------------------------
__global__ __launch_bounds__(32) void mps_chain(const float* __restrict__ x,
                                                const float* __restrict__ tensor,
                                                const unsigned* __restrict__ TLs,
                                                const unsigned* __restrict__ TRs,
                                                float* __restrict__ AlBuf,
                                                float* __restrict__ ArBuf) {
    __shared__ float Asm[16][DB + 1];   // env slice, padded vs bank conflicts

    const int lane = threadIdx.x;
    const int half = lane >> 4;
    const int n    = lane & 15;
    const bool left = (blockIdx.x < 16);
    const int tile  = left ? blockIdx.x : (blockIdx.x - 16);
    const unsigned* Tab = left ? TLs : TRs;

    // ---- boundary init: left: Al0[b,r] = sum_k x[0,b,k]*tensor[0,0,r,k]
    //                     right: Ar0[b,l] = sum_k x[255,b,k]*tensor[255,l,0,k]
    v8f acc[2];
    {
        const int site = left ? 0 : (N_SITES - 1);
        const float* tb = tensor + (size_t)site * DB * DB * DLOC;
        #pragma unroll
        for (int t = 0; t < 2; ++t) {
            #pragma unroll
            for (int v = 0; v < 8; ++v) {
                const int mm  = v + 8 * half;
                const int col = 16 * t + n;
                const int bb  = tile * 16 + mm;
                const float* xr = x + ((size_t)site * BATCH + bb) * DLOC;
                const float* tp = left ? (tb + col * DLOC)           // [0][col][k]
                                       : (tb + (col * DB) * DLOC);   // [col][0][k]
                float s = 0.f;
                #pragma unroll
                for (int k = 0; k < DLOC; ++k) s += xr[k] * tp[k];
                acc[t][v] = s;
            }
        }
    }

    const int m = n;                       // this lane's A-matrix row
    const int b = tile * 16 + m;

    auto siteOf = [&](int s) { return left ? (1 + s) : (254 - s); };

    auto loadB = [&](v16bf Bop[4][2], int s) {
        const unsigned* sb = Tab + (size_t)siteOf(s) * 2048;  // 4*2*32*8 dw/site
        #pragma unroll
        for (int c = 0; c < 4; ++c)
            #pragma unroll
            for (int t = 0; t < 2; ++t)
                Bop[c][t] = *(const v16bf*)(sb + ((c * 2 + t) * 32 + lane) * 8);
    };

    auto prefetchB = [&](int s) {          // pre-warm WGP cache 2 sites ahead
        if (s < 127) {
            const char* p = (const char*)(Tab + (size_t)siteOf(s) * 2048);
            __builtin_prefetch(p + lane * 256, 0, 1);        // 8KB / 32 lanes
            __builtin_prefetch(p + lane * 256 + 128, 0, 1);
        }
    };

    auto step = [&](v16bf Bop[4][2], int s) {
        __syncthreads();                   // prior-iteration LDS reads done
        #pragma unroll
        for (int t = 0; t < 2; ++t)
            #pragma unroll
            for (int v = 0; v < 8; ++v)
                Asm[v + 8 * half][16 * t + n] = acc[t][v];
        __syncthreads();

        const int site = siteOf(s);
        float4 xv = *(const float4*)(x + ((size_t)site * BATCH + b) * DLOC);
        float xk[4] = {xv.x, xv.y, xv.z, xv.w};

        v8f nacc0 = {0.f, 0.f, 0.f, 0.f, 0.f, 0.f, 0.f, 0.f};
        v8f nacc1 = {0.f, 0.f, 0.f, 0.f, 0.f, 0.f, 0.f, 0.f};
        #pragma unroll
        for (int c = 0; c < 4; ++c) {
            // Build U chunk in A-matrix 16x32 bf16 layout:
            // half=0: K = 32c + {0..15}; half=1: K = 32c + {16..31}
            v16bf U;
            #pragma unroll
            for (int j = 0; j < 8; ++j) {
                const int jj = (j < 4) ? j : (j - 4);
                const int l  = 8 * c + 4 * half + (j >> 1);
                const int k0 = (j & 1) * 2;             // compile-time per j
                const float av = Asm[m][l];
                U[2 * j]     = (__bf16)(av * xk[k0]);      // native cvt path
                U[2 * j + 1] = (__bf16)(av * xk[k0 + 1]);
            }
            nacc0 = __builtin_amdgcn_wmma_f32_16x16x32_bf16(
                        false, U, false, Bop[c][0], (short)0, nacc0, false, false);
            nacc1 = __builtin_amdgcn_wmma_f32_16x16x32_bf16(
                        false, U, false, Bop[c][1], (short)0, nacc1, false, false);
        }
        acc[0] = nacc0;
        acc[1] = nacc1;
    };

    // ---- 127 sequential GEMM steps, ping-pong pipelined B-operand loads
    v16bf BA[4][2], BB[4][2];
    loadB(BA, 0);
    prefetchB(1);
    for (int s = 0; s < 127; ) {
        if (s + 1 < 127) { loadB(BB, s + 1); prefetchB(s + 2); }
        step(BA, s);
        ++s;
        if (s >= 127) break;
        if (s + 1 < 127) { loadB(BA, s + 1); prefetchB(s + 2); }
        step(BB, s);
        ++s;
    }

    // ---- write final environment slice
    float* Out = left ? AlBuf : ArBuf;
    #pragma unroll
    for (int t = 0; t < 2; ++t)
        #pragma unroll
        for (int v = 0; v < 8; ++v)
            Out[(size_t)(tile * 16 + v + 8 * half) * DB + 16 * t + n] = acc[t][v];
}

// ---------------------------------------------------------------------------
// Final sandwich: out[b,o] = sum_{l,r} Al[b,l] * Aout[o,l,r] * Ar[b,r]
// ---------------------------------------------------------------------------
__global__ __launch_bounds__(256) void mps_final(const float* __restrict__ Al,
                                                 const float* __restrict__ Ar,
                                                 const float* __restrict__ Aout,
                                                 float* __restrict__ out) {
    int tid = blockIdx.x * 256 + threadIdx.x;
    if (tid >= BATCH * NCLS) return;
    int b = tid / NCLS, o = tid % NCLS;
    const float* al = Al + (size_t)b * DB;
    const float* ar = Ar + (size_t)b * DB;
    const float* ao = Aout + (size_t)o * DB * DB;
    float s = 0.f;
    #pragma unroll 4
    for (int l = 0; l < DB; ++l) {
        float t = 0.f;
        #pragma unroll
        for (int r = 0; r < DB; ++r) t = fmaf(ao[l * DB + r], ar[r], t);
        s = fmaf(al[l], t, s);
    }
    out[tid] = s;
}

extern "C" void kernel_launch(void* const* d_in, const int* in_sizes, int n_in,
                              void* d_out, int out_size, void* d_ws, size_t ws_size,
                              hipStream_t stream) {
    const float* x      = (const float*)d_in[0];  // [256,256,4]
    const float* tensor = (const float*)d_in[1];  // [256,32,32,4]
    const float* Aout   = (const float*)d_in[2];  // [10,32,32]
    float* out = (float*)d_out;                   // [256,10]

    char* ws = (char*)d_ws;
    unsigned* TLs = (unsigned*)(ws);
    unsigned* TRs = (unsigned*)(ws + (size_t)2 * 1024 * 1024);
    float* AlBuf  = (float*)(ws + (size_t)4 * 1024 * 1024);
    float* ArBuf  = (float*)(ws + (size_t)4 * 1024 * 1024 + 32 * 1024);

    // 1) swizzle tensor into bf16 WMMA B-operand tables (2^20 dwords total)
    mps_prep<<<4096, 256, 0, stream>>>(tensor, TLs, TRs);
    // 2) left + right chains, 32 independent waves, 127 WMMA-GEMM steps each
    mps_chain<<<32, 32, 0, stream>>>(x, tensor, TLs, TRs, AlBuf, ArBuf);
    // 3) class-tensor sandwich
    mps_final<<<(BATCH * NCLS + 255) / 256, 256, 0, stream>>>(AlBuf, ArBuf, Aout, out);
}
